// Attention_30588757082379
// MI455X (gfx1250) — matrix operand order repaired and checked by reading, not compiled, not run on hardware
//
#include <hip/hip_runtime.h>
#include <hip/hip_bf16.h>
#include <math.h>

typedef __attribute__((ext_vector_type(16))) __bf16 v16bf;
typedef __attribute__((ext_vector_type(8)))  __bf16 v8bf;
typedef __attribute__((ext_vector_type(8)))  float  v8f;

#define DEC_H 1024
#define ENC2  1024    // 2*ENC_H
#define S_LEN 2048
#define BATCH 32

#if __has_builtin(__builtin_amdgcn_tanhf)
#define FAST_TANH(x) __builtin_amdgcn_tanhf(x)
#else
#define FAST_TANH(x) tanhf(x)
#endif

// ---------------------------------------------------------------------------
// Kernel 1: prepack W_e (= attn_w[:, 1024:2048]) into bf16 in the exact
// per-lane B-operand layout this file uses (32x16 tile, contiguous k per half):
//   lane l<16 : N = n0 + l,      K = k0 + 0..15 (sequential, packed 2/VGPR)
//   lane l>=16: N = n0 + (l-16), K = k0 + 16..31
// Storage: pb[ ((nb*32 + kb)*32 + lane) * 16 + e ]
// ---------------------------------------------------------------------------
__global__ void prepack_B_kernel(const float* __restrict__ attn_w,
                                 __bf16* __restrict__ pb) {
    int t    = blockIdx.x * blockDim.x + threadIdx.x;  // 65536 threads
    int lane = t & 31;
    int kb   = (t >> 5) & 31;
    int nb   = t >> 10;
    int n     = nb * 16 + (lane & 15);
    int kbase = kb * 32 + ((lane & 16) ? 16 : 0);
    const float* src = attn_w + (size_t)n * (DEC_H + ENC2) + DEC_H + kbase;
    __bf16* dst = pb + (size_t)t * 16;
#pragma unroll
    for (int e = 0; e < 16; ++e) dst[e] = (__bf16)src[e];
}

// ---------------------------------------------------------------------------
// Kernel 2: proj_s[b,h] = sum_d s[b,d] * W_s[h,d]   (W_s = attn_w[:, :1024])
// ---------------------------------------------------------------------------
__global__ void proj_s_kernel(const float* __restrict__ s,
                              const float* __restrict__ attn_w,
                              float* __restrict__ proj_s) {
    int t = blockIdx.x * blockDim.x + threadIdx.x;     // 32768 threads
    int b = t >> 10;
    int h = t & 1023;
    const float* srow = s + (size_t)b * DEC_H;
    const float* wrow = attn_w + (size_t)h * (DEC_H + ENC2);
    float acc = 0.0f;
#pragma unroll 8
    for (int d = 0; d < DEC_H; ++d) acc = fmaf(srow[d], wrow[d], acc);
    proj_s[t] = acc;
}

// ---------------------------------------------------------------------------
// Kernel 3: fused GEMM + tanh + v-dot.
// Block = 128 threads (4 waves); M-tile = 64 rows staged once in LDS as bf16.
// LDS layout [l15][sub][k]: per-l15 stride 8208B (8192 + 16 pad -> banks
// spread, dword stride 2052 === 4 mod 64), sub stride 2048B. All A-fragment
// addresses for all 4 subs x 32 k-steps are immediates off ONE base VGPR.
// Each wave owns 16 N-tiles, processed 2 at a time so each A fragment feeds
// two WMMAs (halves DS traffic/wmma); each B fetch is amortized over 64 rows
// (4x less L2 traffic than a 16-row tile).
// ---------------------------------------------------------------------------
__global__ void __launch_bounds__(128)
bahdanau_main_kernel(const float*  __restrict__ enc,     // [32][2048][1024]
                     const __bf16* __restrict__ pb,      // prepacked W_e
                     const float*  __restrict__ proj_s,  // [32][1024]
                     const float*  __restrict__ v_w,     // [1024]
                     float*        __restrict__ scores)  // [32][2048]
{
    constexpr int L15_STRIDE = 4104;                     // bf16 units (8208 B)
    __shared__ __align__(16) __bf16 lA[16 * L15_STRIDE]; // ~128.25 KB
    __shared__ float sred[4][64];

    const int blk  = blockIdx.x;                          // 0..1023
    const int b    = blk >> 5;                            // 32 64-row tiles per batch
    const int s0   = (blk & 31) * 64;
    const int tid  = threadIdx.x;
    const int w    = tid >> 5;                            // wave 0..3
    const int lane = tid & 31;
    const int half = lane >> 4;                           // 0 | 1
    const int l15  = lane & 15;

    // ---- stage A tile: 64 rows x 1024 f32 -> bf16 LDS
    // row r lives at lA[(r&15)*L15_STRIDE + (r>>4)*1024 + k]
    const float* arow = enc + ((size_t)b * S_LEN + s0) * ENC2;
    for (int i = tid; i < 64 * (ENC2 / 4); i += 128) {
        int r  = i >> 8;                                  // row 0..63
        int c4 = i & 255;                                 // float4 index
        float4 v = ((const float4*)(arow + (size_t)r * ENC2))[c4];
        __bf16* dst = lA + (r & 15) * L15_STRIDE + (r >> 4) * 1024 + c4 * 4;
        dst[0] = (__bf16)v.x; dst[1] = (__bf16)v.y;
        dst[2] = (__bf16)v.z; dst[3] = (__bf16)v.w;
    }
    __syncthreads();

    const float* psrow = proj_s + (size_t)b * DEC_H;

    // single A base pointer; everything else is DS immediate offsets
    const __bf16* aBase = lA + l15 * L15_STRIDE + half * 16;

    float sp[4][8];
#pragma unroll
    for (int sub = 0; sub < 4; ++sub)
#pragma unroll
        for (int r = 0; r < 8; ++r) sp[sub][r] = 0.0f;

    for (int nt = 0; nt < 16; nt += 2) {
        const int nb0 = w * 16 + nt;                      // N-tile pair
        v8f acc[2][4] = {};
        // B pointer for nb0; nb1 accesses are +32KB immediates
        const __bf16* bptr = pb + (((size_t)nb0 * 32) * 32 + lane) * 16;
#pragma unroll 2
        for (int kb = 0; kb < 32; ++kb) {
            v16bf bm0 = *(const v16bf*)(bptr + (size_t)kb * 512);
            v16bf bm1 = *(const v16bf*)(bptr + (size_t)kb * 512 + 16384);
#pragma unroll
            for (int sub = 0; sub < 4; ++sub) {
                // A frag matches B prepack: lane<16 -> K{k0..+15},
                // lane>=16 -> K{k0+16..+31} (folded into aBase via half*16)
                const __bf16* aB = aBase + sub * 1024 + kb * 32;
                v8bf a_lo = *(const v8bf*)(aB);
                v8bf a_hi = *(const v8bf*)(aB + 8);
                v16bf a = __builtin_shufflevector(a_lo, a_hi,
                            0,1,2,3,4,5,6,7,8,9,10,11,12,13,14,15);
                acc[0][sub] = __builtin_amdgcn_wmma_f32_16x16x32_bf16(
                                  false, a, false, bm0, (short)0, acc[0][sub],
                                  false, false);
                acc[1][sub] = __builtin_amdgcn_wmma_f32_16x16x32_bf16(
                                  false, a, false, bm1, (short)0, acc[1][sub],
                                  false, false);
            }
        }
        // Epilogue: C/D layout -> elem r is row (sub*16 + r + 8*half),
        // h = (nb0+p)*16 + l15
#pragma unroll
        for (int p = 0; p < 2; ++p) {
            const int   h  = (nb0 + p) * 16 + l15;
            const float ps = psrow[h];
            const float vw = v_w[h];
#pragma unroll
            for (int sub = 0; sub < 4; ++sub)
#pragma unroll
                for (int r = 0; r < 8; ++r)
                    sp[sub][r] += vw * FAST_TANH(acc[p][sub][r] + ps);
        }
    }

    // reduce over the 16 h-columns held by each 16-lane half
#pragma unroll
    for (int sub = 0; sub < 4; ++sub)
#pragma unroll
        for (int r = 0; r < 8; ++r) {
            float v = sp[sub][r];
            for (int m = 1; m < 16; m <<= 1) v += __shfl_xor(v, m, 32);
            sp[sub][r] = v;
        }
    if (l15 == 0) {
#pragma unroll
        for (int sub = 0; sub < 4; ++sub)
#pragma unroll
            for (int r = 0; r < 8; ++r)
                sred[w][sub * 16 + half * 8 + r] = sp[sub][r];
    }
    __syncthreads();

    // sum partial scores across the 4 waves (they split the h range)
    if (tid < 64) {
        float v = sred[0][tid] + sred[1][tid] + sred[2][tid] + sred[3][tid];
        scores[(size_t)b * S_LEN + s0 + tid] = v;
    }
}

// ---------------------------------------------------------------------------
// Kernel 4: softmax over s (length 2048) per batch row.
// ---------------------------------------------------------------------------
__global__ void softmax_kernel(const float* __restrict__ scores,
                               float* __restrict__ out) {
    __shared__ float red[256];
    int b = blockIdx.x;
    int t = threadIdx.x;
    const float* row = scores + (size_t)b * S_LEN;

    float m = -INFINITY;
    for (int i = t; i < S_LEN; i += 256) m = fmaxf(m, row[i]);
    red[t] = m; __syncthreads();
    for (int s = 128; s > 0; s >>= 1) {
        if (t < s) red[t] = fmaxf(red[t], red[t + s]);
        __syncthreads();
    }
    m = red[0]; __syncthreads();

    float sum = 0.0f;
    for (int i = t; i < S_LEN; i += 256) sum += expf(row[i] - m);
    red[t] = sum; __syncthreads();
    for (int s = 128; s > 0; s >>= 1) {
        if (t < s) red[t] += red[t + s];
        __syncthreads();
    }
    float inv = 1.0f / red[0];

    for (int i = t; i < S_LEN; i += 256)
        out[(size_t)b * S_LEN + i] = expf(row[i] - m) * inv;
}

// ---------------------------------------------------------------------------
extern "C" void kernel_launch(void* const* d_in, const int* in_sizes, int n_in,
                              void* d_out, int out_size, void* d_ws, size_t ws_size,
                              hipStream_t stream) {
    const float* s      = (const float*)d_in[0];   // [32][1024]
    const float* enc    = (const float*)d_in[1];   // [32][2048][1024]
    const float* attn_w = (const float*)d_in[2];   // [1024][2048]
    const float* v_w    = (const float*)d_in[3];   // [1][1024]
    float* out          = (float*)d_out;           // [32][2048]

    // workspace layout
    char* ws = (char*)d_ws;
    __bf16* pb     = (__bf16*)ws;                              // 2 MB
    float*  projs  = (float*)(ws + (size_t)2 * 1024 * 1024);   // 128 KB
    float*  scores = (float*)(ws + (size_t)2 * 1024 * 1024 + 128 * 1024); // 256 KB

    prepack_B_kernel<<<256, 256, 0, stream>>>(attn_w, pb);
    proj_s_kernel  <<<128, 256, 0, stream>>>(s, attn_w, projs);
    bahdanau_main_kernel<<<1024, 128, 0, stream>>>(enc, pb, projs, v_w, scores);
    softmax_kernel <<<BATCH, 256, 0, stream>>>(scores, out);
}
